// ForceDime_23390391894194
// MI455X (gfx1250) — matrix operand order repaired and checked by reading, not compiled, not run on hardware
//
#include <hip/hip_runtime.h>
#include <hip/hip_bf16.h>
#include <cmath>

// ---------------- Problem constants (match reference) ----------------
constexpr int NATOMS  = 5000;
constexpr int NEDGES  = 50000;
constexpr int NANGLES = 80000;
constexpr int EMBED   = 128;
constexpr int NRBF    = 6;
constexpr int NSPH    = 6;
constexpr int LSPH    = 7;
constexpr int NBIL    = 8;
constexpr int NCONV   = 3;
constexpr float CUTOFF = 5.0f;

typedef _Float16 v8h  __attribute__((ext_vector_type(8)));
typedef _Float16 v16h __attribute__((ext_vector_type(16)));
typedef float    v8f  __attribute__((ext_vector_type(8)));
typedef int      v4i  __attribute__((ext_vector_type(4)));

struct BesselConst { float z[42]; float norm[42]; };

// Async copy to LDS (gfx1250 GLOBAL_LOAD_ASYNC_TO_LDS_B128, ASYNCcnt-tracked).
// Guarded so the file still compiles if this toolchain doesn't declare the builtin.
// Probe-verified param types: (int4 AS1*, int4 AS3*, imm offset, imm cpol).
#if defined(__HIP_DEVICE_COMPILE__) && __has_builtin(__builtin_amdgcn_global_load_async_to_lds_b128)
#define USE_ASYNC_LDS 1
#else
#define USE_ASYNC_LDS 0
#endif

typedef __attribute__((address_space(1))) v4i* gas_v4i;
typedef __attribute__((address_space(3))) v4i* las_v4i;

__device__ __forceinline__ void stage16B_to_lds(const void* gsrc, void* ldst) {
#if USE_ASYNC_LDS
  __builtin_amdgcn_global_load_async_to_lds_b128((gas_v4i)gsrc, (las_v4i)ldst, 0, 0);
#else
  *(v8h*)ldst = *(const v8h*)gsrc;
#endif
}
__device__ __forceinline__ void wait_async_lds() {
#if USE_ASYNC_LDS
#if __has_builtin(__builtin_amdgcn_s_wait_asynccnt)
  __builtin_amdgcn_s_wait_asynccnt(0);
#else
  asm volatile("s_wait_asynccnt 0x0" ::: "memory");
#endif
#endif
}

// ---------------- Device helpers ----------------
__device__ __forceinline__ float swish_f(float x) { return x / (1.0f + expf(-x)); }
__device__ __forceinline__ float ssp_f(float x) {
  return fmaxf(x, 0.0f) + log1pf(expf(-fabsf(x))) - 0.69314718055994531f;
}
__device__ __forceinline__ float sphjn_d(int l, float x) {
  float s = sinf(x), c = cosf(x);
  float j0 = s / x;
  if (l == 0) return j0;
  float jm = j0, jc = s / (x * x) - c / x;
  for (int ll = 1; ll < l; ++ll) { float nx = (2.0f * ll + 1.0f) / x * jc - jm; jm = jc; jc = nx; }
  return jc;
}
__device__ __forceinline__ v16h cat8(v8h a, v8h b) {
  return __builtin_shufflevector(a, b, 0,1,2,3,4,5,6,7,8,9,10,11,12,13,14,15);
}
// Read a WMMA-A fragment (v16h for one 32-K chunk) for this lane out of an LDS tile row.
__device__ __forceinline__ v16h a_frag_from_lds(const _Float16* Ar, int kc, int hi) {
  v8h lo = *(const v8h*)(Ar + kc * 32 + hi * 16);
  v8h hh = *(const v8h*)(Ar + kc * 32 + hi * 16 + 8);
  return cat8(lo, hh);
}

// ---------------- Weight prep: f32 [k][n] -> f16 [n][k] (WMMA-B friendly) ----------------
// 40 matrices of 128x128: 0-2 W_ji, 3-5 W_kj, 6-8 W_r1, 9-11 W_r2, 12-14 W_fin, 15 ro_W1,
// 16-39 W_bil (layer i, slot b) where src element (e,b,f) is at e*1024 + b*128 + f.
__global__ void prep_weights_kernel(const float* __restrict__ Wji, const float* __restrict__ Wkj,
                                    const float* __restrict__ Wr1, const float* __restrict__ Wr2,
                                    const float* __restrict__ Wfin, const float* __restrict__ roW1,
                                    const float* __restrict__ Wbil, _Float16* __restrict__ WT) {
  int idx = blockIdx.x * 256 + threadIdx.x;
  if (idx >= 40 * 16384) return;
  int mid = idx >> 14;
  int w = idx & 16383;
  int n = w >> 7, k = w & 127;
  float v;
  if (mid < 3)        v = Wji[mid * 16384 + k * 128 + n];
  else if (mid < 6)   v = Wkj[(mid - 3) * 16384 + k * 128 + n];
  else if (mid < 9)   v = Wr1[(mid - 6) * 16384 + k * 128 + n];
  else if (mid < 12)  v = Wr2[(mid - 9) * 16384 + k * 128 + n];
  else if (mid < 15)  v = Wfin[(mid - 12) * 16384 + k * 128 + n];
  else if (mid == 15) v = roW1[k * 128 + n];
  else {
    int t = mid - 16; int li = t >> 3; int b = t & 7;
    v = Wbil[(size_t)li * 131072 + (size_t)k * 1024 + b * 128 + n];
  }
  WT[idx] = (_Float16)v;   // dst layout [mid][n][k] == idx
}

// ---------------- Edge geometry: dvec, d, e_rbf, rad ----------------
__global__ void edge_geom_kernel(const float* __restrict__ nxyz, const int* __restrict__ nbr,
                                 BesselConst bc, float* __restrict__ dvec, float* __restrict__ dist,
                                 float* __restrict__ erbf, float* __restrict__ rad) {
  int e = blockIdx.x * 256 + threadIdx.x;
  if (e >= NEDGES) return;
  int i = nbr[2 * e], j = nbr[2 * e + 1];
  float dx = nxyz[4 * i + 1] - nxyz[4 * j + 1];
  float dy = nxyz[4 * i + 2] - nxyz[4 * j + 2];
  float dz = nxyz[4 * i + 3] - nxyz[4 * j + 3];
  float d = sqrtf(dx * dx + dy * dy + dz * dz);
  dvec[3 * e + 0] = dx; dvec[3 * e + 1] = dy; dvec[3 * e + 2] = dz;
  dist[e] = d;
  float ds = d / CUTOFF;
  float env = 0.0f;
  if (ds < 1.0f) {
    float x5 = ds * ds; x5 *= x5; x5 *= ds;     // ds^5
    env = 1.0f / ds + x5 * (-28.0f + ds * (48.0f - 21.0f * ds));
  }
  const float pref = 0.6324555320336759f;       // sqrt(2/CUTOFF)
  const float PIf = 3.14159265358979323846f;
#pragma unroll
  for (int n = 0; n < NRBF; ++n)
    erbf[(size_t)e * NRBF + n] = pref * env * sinf(PIf * (n + 1) * ds);
  for (int l = 0; l < LSPH; ++l)
    for (int q = 0; q < NSPH; ++q) {
      int c = l * NSPH + q;
      rad[(size_t)e * 42 + c] = bc.norm[c] * sphjn_d(l, bc.z[c] * ds) * env;
    }
}

// ---------------- Angles: Y, a_sbf (fused) -> sb_all[3][NANGLES][8] ----------------
__global__ void angle_kernel(const float* __restrict__ nxyz, const int* __restrict__ angle,
                             const int* __restrict__ kj_idx, const float* __restrict__ rad,
                             const float* __restrict__ Wsbf, float* __restrict__ sb_all) {
  int a = blockIdx.x * 256 + threadIdx.x;
  if (a >= NANGLES) return;
  int ia = angle[3 * a], ja = angle[3 * a + 1], ka = angle[3 * a + 2];
  float rji[3], rjk[3];
#pragma unroll
  for (int c = 0; c < 3; ++c) {
    rji[c] = nxyz[4 * ia + 1 + c] - nxyz[4 * ja + 1 + c];
    rjk[c] = nxyz[4 * ka + 1 + c] - nxyz[4 * ja + 1 + c];
  }
  float cx = rji[1] * rjk[2] - rji[2] * rjk[1];
  float cy = rji[2] * rjk[0] - rji[0] * rjk[2];
  float cz = rji[0] * rjk[1] - rji[1] * rjk[0];
  float cn = sqrtf(cx * cx + cy * cy + cz * cz);
  float dp = rji[0] * rjk[0] + rji[1] * rjk[1] + rji[2] * rjk[2];
  float ca = cosf(atan2f(cn, dp));
  float P[LSPH];
  P[0] = 1.0f; P[1] = ca;
#pragma unroll
  for (int l = 1; l < LSPH - 1; ++l)
    P[l + 1] = ((2 * l + 1) * ca * P[l] - l * P[l - 1]) / (l + 1);
  float Y[LSPH];
  const float inv4pi = 0.07957747154594767f;  // 1/(4*pi)
#pragma unroll
  for (int l = 0; l < LSPH; ++l) Y[l] = sqrtf((2 * l + 1) * inv4pi) * P[l];
  const float* rr = rad + (size_t)kj_idx[a] * 42;
  float ry[42];
  for (int l = 0; l < LSPH; ++l)
    for (int q = 0; q < NSPH; ++q) ry[l * NSPH + q] = rr[l * NSPH + q] * Y[l];
  for (int li = 0; li < NCONV; ++li)
    for (int b = 0; b < NBIL; ++b) {
      float s = 0.0f;
      for (int c = 0; c < 42; ++c) s += ry[c] * Wsbf[li * 336 + c * 8 + b];
      sb_all[((size_t)li * NANGLES + a) * 8 + b] = s;
    }
}

// ---------------- e_rbf @ W (6x128) projection (+optional bias & swish) ----------------
__global__ void rbf_proj_kernel(const float* __restrict__ erbf, const float* __restrict__ W,
                                const float* __restrict__ bias, float* __restrict__ out, int do_swish) {
  int idx = blockIdx.x * 256 + threadIdx.x;      // exactly NEDGES*128 threads
  int e = idx >> 7, n = idx & 127;
  float s = bias ? bias[n] : 0.0f;
#pragma unroll
  for (int q = 0; q < NRBF; ++q) s += erbf[(size_t)e * NRBF + q] * W[q * 128 + n];
  out[idx] = do_swish ? swish_f(s) : s;
}

// ---------------- Embedding: m = swish([h_i | h_j | e_emb] @ W_m + b_m) ----------------
__global__ void embed_kernel(const float* __restrict__ nxyz, const int* __restrict__ nbr,
                             const float* __restrict__ emb, const float* __restrict__ Wm,
                             const float* __restrict__ bm, const float* __restrict__ eemb,
                             float* __restrict__ m, _Float16* __restrict__ mh) {
  int idx = blockIdx.x * 256 + threadIdx.x;      // NEDGES*128 threads
  int e = idx >> 7, n = idx & 127;
  int ia = nbr[2 * e], ja = nbr[2 * e + 1];
  int zi = (int)nxyz[4 * ia], zj = (int)nxyz[4 * ja];
  const float* hi = emb + (size_t)zi * 128;
  const float* hj = emb + (size_t)zj * 128;
  const float* ee = eemb + (size_t)e * 128;
  float s = bm[n];
  for (int k = 0; k < 128; ++k) s += hi[k] * Wm[k * 128 + n];
  for (int k = 0; k < 128; ++k) s += hj[k] * Wm[(128 + k) * 128 + n];
  for (int k = 0; k < 128; ++k) s += ee[k] * Wm[(256 + k) * 128 + n];
  s = swish_f(s);
  m[idx] = s;
  mh[idx] = (_Float16)s;
}

// ---------------- WMMA GEMM: out = epi(A(rows x 128) @ W(128x128) + bias) ----------------
// Block: 256 threads = 8 wave32 waves; block computes 16 rows x 128 cols; wave w -> cols [16w,16w+16).
// A tile (16x128 f16 = 4KB) is staged into LDS once per block (async-to-LDS when available),
// then each wave reads its A fragments from LDS. WT: f16 [n][k] pre-transposed.
// EPI: 0 swish, 1 swish*extra, 2 extra + swish, 3 ssp.
template <int EPI>
__global__ void gemm128_kernel(const _Float16* __restrict__ A, const _Float16* __restrict__ WT,
                               const float* __restrict__ bias, const float* __restrict__ extra,
                               float* __restrict__ outF, _Float16* __restrict__ outH) {
  __shared__ __align__(16) _Float16 smem[16 * 128];
  const int lane = threadIdx.x & 31;
  const int wv = threadIdx.x >> 5;
  const int hi = lane >> 4;
  const int l15 = lane & 15;
  const int m0 = blockIdx.x * 16;
  // ---- cooperative A-tile staging: thread t moves 16B of row (t/16), segment (t%16) ----
  {
    const int srow = threadIdx.x >> 4, sseg = threadIdx.x & 15;
    stage16B_to_lds(A + (size_t)(m0 + srow) * 128 + sseg * 8, &smem[srow * 128 + sseg * 8]);
  }
  if (EPI == 1 || EPI == 2)
    __builtin_prefetch(extra + (size_t)m0 * 128 + wv * 256, 0, 1);   // epilogue rows -> global_prefetch_b8
  wait_async_lds();
  __syncthreads();
  const _Float16* Ar = smem + (size_t)l15 * 128;
  const int c = wv * 16 + l15;                    // output column
  const _Float16* Wc = WT + (size_t)c * 128;      // B operand: lane col = c; K = kc*32 + hi*16 + j
  v8f acc = {};
#pragma unroll
  for (int kc = 0; kc < 4; ++kc) {
    v16h a = a_frag_from_lds(Ar, kc, hi);
    v16h b = *(const v16h*)(Wc + kc * 32 + hi * 16);
    acc = __builtin_amdgcn_wmma_f32_16x16x32_f16(false, a, false, b, (short)0, acc, false, false);
  }
  const float bcol = bias[c];
#pragma unroll
  for (int j = 0; j < 8; ++j) {                   // D layout: row = m0 + j + hi*8, col = c
    const int rj = m0 + j + hi * 8;
    float v = acc[j] + bcol;
    if (EPI == 0)      v = swish_f(v);
    else if (EPI == 1) v = swish_f(v) * extra[(size_t)rj * 128 + c];
    else if (EPI == 2) v = extra[(size_t)rj * 128 + c] + swish_f(v);
    else               v = ssp_f(v);
    if (outF) outF[(size_t)rj * 128 + c] = v;
    if (outH) outH[(size_t)rj * 128 + c] = (_Float16)v;
  }
}

// ---------------- Bilinear message: x[a,f] = sum_b sb[a,b] * (x_kj[kj[a]] @ Wbil[:,b,:])[f],
// scatter-added into xacc[ji_idx[a]] via native f32 L2 atomics ----------------
__global__ void bilinear_kernel(const _Float16* __restrict__ Xh, const _Float16* __restrict__ WBT,
                                const float* __restrict__ sb, const int* __restrict__ kj_idx,
                                const int* __restrict__ ji_idx, float* __restrict__ xacc) {
  __shared__ __align__(16) _Float16 smem[16 * 128];
  const int lane = threadIdx.x & 31;
  const int wv = threadIdx.x >> 5;
  const int hi = lane >> 4;
  const int l15 = lane & 15;
  const int a0 = blockIdx.x * 16;
  // ---- cooperative gathered A-tile staging (rows via kj_idx) ----
  {
    const int srow = threadIdx.x >> 4, sseg = threadIdx.x & 15;
    const int grow = kj_idx[a0 + srow];
    stage16B_to_lds(Xh + (size_t)grow * 128 + sseg * 8, &smem[srow * 128 + sseg * 8]);
  }
  __builtin_prefetch(sb + (size_t)a0 * 8, 0, 1);                     // sb block for epilogue
  wait_async_lds();
  __syncthreads();
  const _Float16* Ar = smem + (size_t)l15 * 128;
  v16h a[4];
#pragma unroll
  for (int kc = 0; kc < 4; ++kc) a[kc] = a_frag_from_lds(Ar, kc, hi);
  const int c = wv * 16 + l15;
  float accv[8] = {0.f, 0.f, 0.f, 0.f, 0.f, 0.f, 0.f, 0.f};
#pragma unroll
  for (int b = 0; b < NBIL; ++b) {
    const _Float16* Wc = WBT + (size_t)b * 16384 + (size_t)c * 128;
    v8f s = {};
#pragma unroll
    for (int kc = 0; kc < 4; ++kc) {
      v16h bb = *(const v16h*)(Wc + kc * 32 + hi * 16);
      s = __builtin_amdgcn_wmma_f32_16x16x32_f16(false, a[kc], false, bb, (short)0, s, false, false);
    }
#pragma unroll
    for (int j = 0; j < 8; ++j)
      accv[j] += sb[(size_t)(a0 + j + hi * 8) * 8 + b] * s[j];
  }
#pragma unroll
  for (int j = 0; j < 8; ++j) {
    const int arj = a0 + j + hi * 8;
    const int eo = ji_idx[arj];
    unsafeAtomicAdd(&xacc[(size_t)eo * 128 + c], accv[j]);
  }
}

// ---------------- Misc small kernels ----------------
__global__ void f2h_kernel(const float* __restrict__ src, _Float16* __restrict__ dst) {
  int idx = blockIdx.x * 256 + threadIdx.x;      // NEDGES*128 threads
  dst[idx] = (_Float16)src[idx];
}
__global__ void zero_kernel(float* p, int n) {
  int i = blockIdx.x * 256 + threadIdx.x;
  if (i < n) p[i] = 0.0f;
}
__global__ void force_kernel(const float* __restrict__ t, const float* __restrict__ roW2,
                             const float* __restrict__ rob2, const int* __restrict__ nbr,
                             const float* __restrict__ dvec, float* __restrict__ fout) {
  int e = blockIdx.x * 256 + threadIdx.x;
  if (e >= NEDGES) return;
  float s = rob2[0];
  const float* te = t + (size_t)e * 128;
  for (int n = 0; n < 128; ++n) s += te[n] * roW2[n];
  int ia = nbr[2 * e], ja = nbr[2 * e + 1];
#pragma unroll
  for (int c = 0; c < 3; ++c) {
    float f = s * dvec[3 * e + c];
    unsafeAtomicAdd(&fout[3 * ia + c], f);
    unsafeAtomicAdd(&fout[3 * ja + c], -f);
  }
}

// ---------------- Host-side Bessel constants (match reference numpy code) ----------------
static double sphjn_h(int l, double x) {
  double j0 = sin(x) / x;
  if (l == 0) return j0;
  double jm = j0, jc = sin(x) / (x * x) - cos(x) / x;
  for (int ll = 1; ll < l; ++ll) { double nx = (2.0 * ll + 1.0) / x * jc - jm; jm = jc; jc = nx; }
  return jc;
}
static void compute_bessel(BesselConst* bc) {
  const int nz = NSPH + LSPH;                    // 13
  double zeros[LSPH][13];
  for (int n = 0; n < nz; ++n) zeros[0][n] = (n + 1) * M_PI;
  for (int l = 1; l < LSPH; ++l) {
    for (int n = 0; n < nz - l; ++n) {
      double a = zeros[l - 1][n], b = zeros[l - 1][n + 1];
      double fa = sphjn_h(l, a);
      for (int it = 0; it < 80; ++it) {
        double mid = 0.5 * (a + b);
        double fm = sphjn_h(l, mid);
        if ((fm > 0) == (fa > 0)) { a = mid; fa = fm; } else { b = mid; }
      }
      zeros[l][n] = 0.5 * (a + b);
    }
  }
  for (int l = 0; l < LSPH; ++l)
    for (int n = 0; n < NSPH; ++n) {
      bc->z[l * NSPH + n] = (float)zeros[l][n];
      bc->norm[l * NSPH + n] = (float)(sqrt(2.0) / fabs(sphjn_h(l + 1, zeros[l][n])));
    }
}

// ---------------- Workspace layout (bytes) ----------------
constexpr size_t OFF_WT   = 0;                                   // 40*16384 f16 = 1,310,720 B
constexpr size_t OFF_D    = OFF_WT + 40 * 16384 * 2;             // 200,000
constexpr size_t OFF_DVEC = OFF_D + (size_t)NEDGES * 4;          // 600,000
constexpr size_t OFF_ERBF = OFF_DVEC + (size_t)NEDGES * 12;      // 1,200,000
constexpr size_t OFF_RAD  = OFF_ERBF + (size_t)NEDGES * 24;      // 8,400,000
constexpr size_t OFF_SB   = OFF_RAD + (size_t)NEDGES * 168;      // 7,680,000
constexpr size_t OFF_BUFA = OFF_SB + (size_t)NCONV * NANGLES * 32;
constexpr size_t OFF_M    = OFF_BUFA + (size_t)NEDGES * 512;
constexpr size_t OFF_XACC = OFF_M + (size_t)NEDGES * 512;
constexpr size_t OFF_MH   = OFF_XACC + (size_t)NEDGES * 512;
constexpr size_t OFF_XKJH = OFF_MH + (size_t)NEDGES * 256;
constexpr size_t OFF_HC   = OFF_XKJH + (size_t)NEDGES * 256;
constexpr size_t WS_NEEDED = OFF_HC + (size_t)NEDGES * 256;      // ~134.6 MB (L2-resident)

extern "C" void kernel_launch(void* const* d_in, const int* in_sizes, int n_in,
                              void* d_out, int out_size, void* d_ws, size_t ws_size,
                              hipStream_t stream) {
  if (ws_size < WS_NEEDED) return;
  // ----- inputs (setup_inputs dict order; params flattened in insertion order) -----
  const float* nxyz    = (const float*)d_in[0];
  const int*   nbr     = (const int*)d_in[1];
  const int*   angle   = (const int*)d_in[2];
  const int*   kj_idx  = (const int*)d_in[3];
  const int*   ji_idx  = (const int*)d_in[4];
  const float* atom_emb = (const float*)d_in[6];
  const float* emb_We  = (const float*)d_in[7];
  const float* emb_be  = (const float*)d_in[8];
  const float* emb_Wm  = (const float*)d_in[9];
  const float* emb_bm  = (const float*)d_in[10];
  const float* W_ji    = (const float*)d_in[11];
  const float* b_ji    = (const float*)d_in[12];
  const float* W_kj    = (const float*)d_in[13];
  const float* b_kj    = (const float*)d_in[14];
  const float* W_rbf   = (const float*)d_in[15];
  const float* W_sbf   = (const float*)d_in[16];
  const float* W_bil   = (const float*)d_in[17];
  const float* W_r1    = (const float*)d_in[18];
  const float* b_r1    = (const float*)d_in[19];
  const float* W_r2    = (const float*)d_in[20];
  const float* b_r2    = (const float*)d_in[21];
  const float* W_fin   = (const float*)d_in[22];
  const float* b_fin   = (const float*)d_in[23];
  const float* ro_W1   = (const float*)d_in[24];
  const float* ro_b1   = (const float*)d_in[25];
  const float* ro_W2   = (const float*)d_in[26];
  const float* ro_b2   = (const float*)d_in[27];
  float* fout = (float*)d_out;

  char* ws = (char*)d_ws;
  _Float16* WT   = (_Float16*)(ws + OFF_WT);
  float* dist    = (float*)(ws + OFF_D);
  float* dvec    = (float*)(ws + OFF_DVEC);
  float* erbf    = (float*)(ws + OFF_ERBF);
  float* rad     = (float*)(ws + OFF_RAD);
  float* sb_all  = (float*)(ws + OFF_SB);
  float* bufA    = (float*)(ws + OFF_BUFA);   // e_emb -> rbfp -> readout t
  float* mbuf    = (float*)(ws + OFF_M);
  float* xacc    = (float*)(ws + OFF_XACC);
  _Float16* mh   = (_Float16*)(ws + OFF_MH);
  _Float16* xkjh = (_Float16*)(ws + OFF_XKJH);
  _Float16* hC   = (_Float16*)(ws + OFF_HC);

  BesselConst bc;
  compute_bessel(&bc);

  const int EB = (NEDGES + 255) / 256;
  const int AB = (NANGLES + 255) / 256;
  const int FB = NEDGES * 128 / 256;        // 25000, exact
  const int GT = NEDGES / 16;               // 3125 GEMM tiles
  const int BT = NANGLES / 16;              // 5000 bilinear tiles

  // 1. weight prep (f32 [k][n] -> f16 [n][k])
  prep_weights_kernel<<<(40 * 16384 + 255) / 256, 256, 0, stream>>>(
      W_ji, W_kj, W_r1, W_r2, W_fin, ro_W1, W_bil, WT);
  // 2. geometry + basis
  edge_geom_kernel<<<EB, 256, 0, stream>>>(nxyz, nbr, bc, dvec, dist, erbf, rad);
  angle_kernel<<<AB, 256, 0, stream>>>(nxyz, angle, kj_idx, rad, W_sbf, sb_all);
  // 3. embedding
  rbf_proj_kernel<<<FB, 256, 0, stream>>>(erbf, emb_We, emb_be, bufA, 1);     // e_emb
  embed_kernel<<<FB, 256, 0, stream>>>(nxyz, nbr, atom_emb, emb_Wm, emb_bm, bufA, mbuf, mh);
  // 4. interaction layers
  for (int i = 0; i < NCONV; ++i) {
    const _Float16* WTji  = WT + (size_t)(0 + i) * 16384;
    const _Float16* WTkj  = WT + (size_t)(3 + i) * 16384;
    const _Float16* WTr1  = WT + (size_t)(6 + i) * 16384;
    const _Float16* WTr2  = WT + (size_t)(9 + i) * 16384;
    const _Float16* WTfin = WT + (size_t)(12 + i) * 16384;
    const _Float16* WBT   = WT + (size_t)(16 + i * 8) * 16384;
    rbf_proj_kernel<<<FB, 256, 0, stream>>>(erbf, W_rbf + (size_t)i * 768, nullptr, bufA, 0);
    // x_ji -> xacc (f32)
    gemm128_kernel<0><<<GT, 256, 0, stream>>>(mh, WTji, b_ji + i * 128, nullptr, xacc, nullptr);
    // x_kj = swish(m@Wkj+b)*rbfp -> f16 only
    gemm128_kernel<1><<<GT, 256, 0, stream>>>(mh, WTkj, b_kj + i * 128, bufA, nullptr, xkjh);
    // bilinear message, scatter-add into xacc
    bilinear_kernel<<<BT, 256, 0, stream>>>(xkjh, WBT, sb_all + (size_t)i * NANGLES * 8,
                                            kj_idx, ji_idx, xacc);
    // residual MLP
    f2h_kernel<<<FB, 256, 0, stream>>>(xacc, hC);                                   // out -> f16
    gemm128_kernel<0><<<GT, 256, 0, stream>>>(hC, WTr1, b_r1 + i * 128, nullptr, nullptr, xkjh);
    gemm128_kernel<2><<<GT, 256, 0, stream>>>(xkjh, WTr2, b_r2 + i * 128, xacc, nullptr, hC);
    // m = m + swish(out@Wfin+b) (element-wise in-place on m is safe)
    gemm128_kernel<2><<<GT, 256, 0, stream>>>(hC, WTfin, b_fin + i * 128, mbuf, mbuf, mh);
  }
  // 5. readout
  gemm128_kernel<3><<<GT, 256, 0, stream>>>(mh, WT + (size_t)15 * 16384, ro_b1, nullptr, bufA, nullptr);
  zero_kernel<<<(NATOMS * 3 + 255) / 256, 256, 0, stream>>>(fout, NATOMS * 3);
  force_kernel<<<EB, 256, 0, stream>>>(bufA, ro_W2, ro_b2, nbr, dvec, fout);
}
